// DG_Interaction_45561013076174
// MI455X (gfx1250) — matrix operand order repaired and checked by reading, not compiled, not run on hardware
//
#include <hip/hip_runtime.h>
#include <hip/hip_bf16.h>

// ---------------------------------------------------------------------------
// MI455X (gfx1250, wave32) fused GNN pipeline.
//   - per-edge scatter-add with f32 global atomics (HBM-bound part, ~4.8 GB ->
//     ~0.2 ms at 23.3 TB/s; this dominates the roofline)
//   - all GEMMs via v_wmma_f32_16x16x32_f16 (f16 in, f32 accum), BM=32 rows
//     per block, B fragments double-buffered, weights L2-resident (<370 KB)
//   - f16 A tiles staged via GLOBAL_LOAD_ASYNC_TO_LDS_B128 (ASYNCcnt path)
//   - bias/ReLU/LayerNorm fused into the GEMM epilogue
// ---------------------------------------------------------------------------

typedef __attribute__((ext_vector_type(16))) _Float16 v16h;
typedef __attribute__((ext_vector_type(8)))  _Float16 v8h;
typedef __attribute__((ext_vector_type(8)))  float    v8f;

#define GN_N 100000
#define GN_E 1000000
#define GN_D 300
#define NPAD 304      // output cols padded to 19 tiles of 16
#define NTILES 19
#define BM 32         // rows per block (2 row-tiles of 16)

// ---------------------------------------------------------------------------
// Weight prep: W [Ksrc x 300] (row major) -> Wt [304 x Kpad] f16, transposed,
// zero padded.  Wt[n*Kpad + k] = W[k*300 + n].
// ---------------------------------------------------------------------------
__global__ void prep_wt_kernel(const float* __restrict__ W, _Float16* __restrict__ Wt,
                               int Ksrc, int Kpad) {
  int idx = blockIdx.x * blockDim.x + threadIdx.x;
  int tot = NPAD * Kpad;
  if (idx >= tot) return;
  int n = idx / Kpad;
  int k = idx - n * Kpad;
  float v = 0.0f;
  if (n < GN_D && k < Ksrc) v = W[(size_t)k * GN_D + n];
  Wt[idx] = (_Float16)v;
}

// ---------------------------------------------------------------------------
// Degree counting: edges [2 x E] int32 (src row then dst row).
// ---------------------------------------------------------------------------
__global__ void degree_kernel(const int* __restrict__ edges, int E,
                              int* __restrict__ deg_out, int* __restrict__ deg_in) {
  int e = blockIdx.x * blockDim.x + threadIdx.x;
  if (e >= E) return;
  atomicAdd(&deg_out[edges[e]], 1);
  atomicAdd(&deg_in[edges[E + e]], 1);
}

// norm[i] = deg>0 ? rsqrt(deg) : 0   (4 contiguous arrays of N handled at once)
__global__ void norm_kernel(const int* __restrict__ deg, float* __restrict__ norm, int count) {
  int t = blockIdx.x * blockDim.x + threadIdx.x;
  if (t >= count) return;
  int d = deg[t];
  norm[t] = (d > 0) ? rsqrtf((float)d) : 0.0f;
}

// ---------------------------------------------------------------------------
// Scatter: one wave32 per edge. agg[dst] += x[src] * norm_src[src]
// ---------------------------------------------------------------------------
__global__ void __launch_bounds__(256)
scatter_kernel(const float* __restrict__ x, const int* __restrict__ edges,
               const float* __restrict__ norm_src, float* __restrict__ agg, int E) {
  int gw = (blockIdx.x * blockDim.x + threadIdx.x) >> 5;
  int lane = threadIdx.x & 31;
  if (gw >= E) return;
  int s = edges[gw];
  int d = edges[E + gw];
  float ns = norm_src[s];
  const float* xr = x + (size_t)s * GN_D;
  float* ar = agg + (size_t)d * GN_D;
  for (int c = lane; c < GN_D; c += 32)
    atomicAdd(&ar[c], xr[c] * ns);
}

// ---------------------------------------------------------------------------
// WMMA fragment helpers (wave32 layouts, CDNA5 ISA 7.12.2):
//   A 16x32 f16 : lane l (r=l&15, h=l>>4) holds K = h*8+{0..7}, 16+h*8+{0..7}
//   B 32x16 f16 : lane l (c=l&15, h=l>>4) holds K = h*8+{0..7}, 16+h*8+{0..7} of column c
//   C/D 16x16 f32: elem v of lane l -> row v+8*(l>>4), col l&15
// ---------------------------------------------------------------------------
__device__ __forceinline__ v16h load_a_frag(const _Float16* As, int rowbase, int Kpad,
                                            int c, int half, int k0) {
  const _Float16* pa = As + (size_t)(rowbase + c) * Kpad + k0 + half * 8;
  v8h lo = *(const v8h*)pa;
  v8h hi = *(const v8h*)(pa + 16);
  return __builtin_shufflevector(lo, hi, 0, 1, 2, 3, 4, 5, 6, 7,
                                 8, 9, 10, 11, 12, 13, 14, 15);
}

__device__ __forceinline__ v16h load_b_frag(const _Float16* Wt, int ct, int Kpad,
                                            int c, int half, int k0) {
  const _Float16* pb = Wt + (size_t)(ct * 16 + c) * Kpad + k0 + half * 8;
  v8h lo = *(const v8h*)pb;
  v8h hi = *(const v8h*)(pb + 16);
  return __builtin_shufflevector(lo, hi, 0, 1, 2, 3, 4, 5, 6, 7,
                                 8, 9, 10, 11, 12, 13, 14, 15);
}

#define WMMA_F16(a, b, acc) \
  __builtin_amdgcn_wmma_f32_16x16x32_f16(false, (a), false, (b), (short)0, (acc), false, false)

// async global->LDS copy of 16 bytes per active lane (tracked by ASYNCcnt)
__device__ __forceinline__ void async_ld_b128(unsigned int lds_off, const void* gptr) {
  asm volatile("global_load_async_to_lds_b128 %0, %1, off"
               :: "v"(lds_off), "v"(gptr)
               : "memory");
}
__device__ __forceinline__ void wait_async0() {
#if __has_builtin(__builtin_amdgcn_s_wait_asynccnt)
  __builtin_amdgcn_s_wait_asynccnt(0);
#else
  asm volatile("s_wait_asynccnt 0x0" ::: "memory");
#endif
}

// ---------------------------------------------------------------------------
// Fused GEMM: out = epilogue( A[BM x K] @ Wt^T + bias )
//   block: 32 rows x 304 cols, 8 waves; wave w owns col tiles {w, w+8, w+16}
//   (third tile only for w<3 — wave-uniform, so EXEC stays all-1s per wave),
//   and both 16-row tiles, so each B fragment feeds 2 WMMAs.
//   A staged in LDS as f16: f32 path fuses per-row scale (norm_dst) + convert;
//   f16 path (lda16 == Kpad, pre-padded) is one contiguous async LDS copy.
//   epilogue: ReLU -> f16 [N x 320]  (conv layers)
//          or LayerNorm -> f16 into merged[N x 608] at col offset (support)
//          or LayerNorm -> f32 d_out (merge layer)
// ---------------------------------------------------------------------------
__global__ void __launch_bounds__(256)
gemm_fused_kernel(const float* __restrict__ A32, const float* __restrict__ rowscale,
                  const _Float16* __restrict__ A16, int lda16,
                  int Kvalid, int Kpad,
                  const _Float16* __restrict__ Wt, const float* __restrict__ bias,
                  int do_relu, int do_ln,
                  const float* __restrict__ gamma, const float* __restrict__ beta,
                  _Float16* __restrict__ out16, int ldo16, int coff, int zero_hi,
                  float* __restrict__ out32) {
  // As (accumulate phase) and Out (epilogue) have disjoint lifetimes -> union
  __shared__ __align__(16) union SMem {
    _Float16 As[BM * 608];     // max Kpad = 608  (38,912 B)
    float    Out[BM][308];     // 39,424 B
  } sm;
  __shared__ float red[2][BM][8];
  __shared__ float mu[BM], rsv[BM];

  const int tid = threadIdx.x;
  const int row0 = blockIdx.x * BM;

  // ---- stage A tile into LDS (f16, zero padded to Kpad) ----
  if (A32 != nullptr) {
    const int tot = BM * Kpad;
    for (int idx = tid; idx < tot; idx += 256) {
      int r = idx / Kpad, k = idx - r * Kpad;
      float v = 0.0f;
      if (k < Kvalid) v = A32[(size_t)(row0 + r) * Kvalid + k] * rowscale[row0 + r];
      sm.As[idx] = (_Float16)v;
    }
  } else {
    // contiguous BM*Kpad*2 bytes: async copy straight into LDS
    const char* gbase = (const char*)(A16 + (size_t)row0 * Kpad);
    unsigned int lbase = (unsigned int)(uintptr_t)&sm.As[0];  // low 32 bits = LDS offset
    const int nbytes = BM * Kpad * 2;
    for (int ofs = tid * 16; ofs < nbytes; ofs += 256 * 16)
      async_ld_b128(lbase + (unsigned int)ofs, gbase + ofs);
    wait_async0();
  }
  __syncthreads();

  const int wv = tid >> 5;
  const int lane = tid & 31;
  const int half = lane >> 4;
  const int c = lane & 15;
  const int ct0 = wv, ct1 = wv + 8, ct2 = wv + 16;
  const bool has3 = (ct2 < NTILES);

  const v8f vzero = {0.f, 0.f, 0.f, 0.f, 0.f, 0.f, 0.f, 0.f};
  v8f a00 = vzero, a10 = vzero;   // [row-tile][col-tile]
  v8f a01 = vzero, a11 = vzero;
  v8f a02 = vzero, a12 = vzero;

  // double-buffered B fragments
  v16h b0 = load_b_frag(Wt, ct0, Kpad, c, half, 0);
  v16h b1 = load_b_frag(Wt, ct1, Kpad, c, half, 0);
  v16h b2 = b0;
  if (has3) b2 = load_b_frag(Wt, ct2, Kpad, c, half, 0);

  for (int k0 = 0; k0 < Kpad; k0 += 32) {
    v16h af0 = load_a_frag(sm.As, 0, Kpad, c, half, k0);
    v16h af1 = load_a_frag(sm.As, 16, Kpad, c, half, k0);
    v16h nb0 = b0, nb1 = b1, nb2 = b2;
    const int kn = k0 + 32;
    if (kn < Kpad) {
      nb0 = load_b_frag(Wt, ct0, Kpad, c, half, kn);
      nb1 = load_b_frag(Wt, ct1, Kpad, c, half, kn);
      if (has3) nb2 = load_b_frag(Wt, ct2, Kpad, c, half, kn);
    }
    a00 = WMMA_F16(af0, b0, a00);
    a10 = WMMA_F16(af1, b0, a10);
    a01 = WMMA_F16(af0, b1, a01);
    a11 = WMMA_F16(af1, b1, a11);
    if (has3) {
      a02 = WMMA_F16(af0, b2, a02);
      a12 = WMMA_F16(af1, b2, a12);
    }
    b0 = nb0; b1 = nb1; b2 = nb2;
  }

  __syncthreads();  // everyone done reading As -> safe to reuse as Out

  // ---- bias (+ ReLU), dump accumulators to LDS row-major ----
  {
    auto dump = [&](const v8f& accv, int rt, int col) {
      float bv = (col < GN_D) ? bias[col] : 0.0f;
#pragma unroll
      for (int v = 0; v < 8; v++) {
        float x = accv[v] + bv;
        if (do_relu) x = fmaxf(x, 0.0f);
        sm.Out[rt * 16 + v + 8 * half][col] = x;
      }
    };
    dump(a00, 0, ct0 * 16 + c);
    dump(a10, 1, ct0 * 16 + c);
    dump(a01, 0, ct1 * 16 + c);
    dump(a11, 1, ct1 * 16 + c);
    if (has3) {
      dump(a02, 0, ct2 * 16 + c);
      dump(a12, 1, ct2 * 16 + c);
    }
  }
  __syncthreads();

  if (!do_ln) {
    // conv epilogue: f16 [N x ldo16] with zeroed padding cols
    for (int idx = tid; idx < BM * ldo16; idx += 256) {
      int r = idx / ldo16, cc = idx - r * ldo16;
      float v = (cc < NPAD) ? sm.Out[r][cc] : 0.0f;
      out16[(size_t)(row0 + r) * ldo16 + cc] = (_Float16)v;
    }
    return;
  }

  // ---- LayerNorm over the 300 valid cols of each row ----
  {
    int r = tid >> 3, j = tid & 7;
    float s = 0.0f, sq = 0.0f;
    for (int cc = j; cc < GN_D; cc += 8) {
      float v = sm.Out[r][cc];
      s += v;
      sq += v * v;
    }
    red[0][r][j] = s;
    red[1][r][j] = sq;
  }
  __syncthreads();
  if (tid < BM) {
    float s = 0.0f, sq = 0.0f;
#pragma unroll
    for (int j = 0; j < 8; j++) { s += red[0][tid][j]; sq += red[1][tid][j]; }
    float m = s * (1.0f / GN_D);
    float var = fmaxf(sq * (1.0f / GN_D) - m * m, 0.0f);
    mu[tid] = m;
    rsv[tid] = rsqrtf(var + 1e-5f);
  }
  __syncthreads();
  for (int idx = tid; idx < BM * GN_D; idx += 256) {
    int r = idx / GN_D, cc = idx - r * GN_D;
    float y = (sm.Out[r][cc] - mu[r]) * rsv[r] * gamma[cc] + beta[cc];
    if (out32 != nullptr)
      out32[(size_t)(row0 + r) * GN_D + cc] = y;
    else
      out16[(size_t)(row0 + r) * ldo16 + coff + cc] = (_Float16)y;
  }
  if (zero_hi) {  // zero merged cols 600..607 (K padding of the merge GEMM)
    for (int idx = tid; idx < BM * 8; idx += 256) {
      int r = idx >> 3, cc = idx & 7;
      out16[(size_t)(row0 + r) * ldo16 + 600 + cc] = (_Float16)0.0f;
    }
  }
}

// ---------------------------------------------------------------------------
extern "C" void kernel_launch(void* const* d_in, const int* in_sizes, int n_in,
                              void* d_out, int out_size, void* d_ws, size_t ws_size,
                              hipStream_t stream) {
  const int N = GN_N, E = GN_E;

  const float* x      = (const float*)d_in[0];
  const int*   row_ed = (const int*)d_in[1];
  const int*   col_ed = (const int*)d_in[2];
  const float* Wr  = (const float*)d_in[3];
  const float* br  = (const float*)d_in[4];
  const float* Wc  = (const float*)d_in[5];
  const float* bc  = (const float*)d_in[6];
  const float* Wrs = (const float*)d_in[7];
  const float* brs = (const float*)d_in[8];
  const float* g_rs = (const float*)d_in[9];
  const float* b_rs = (const float*)d_in[10];
  const float* Wcs = (const float*)d_in[11];
  const float* bcs = (const float*)d_in[12];
  const float* g_cs = (const float*)d_in[13];
  const float* b_cs = (const float*)d_in[14];
  const float* Wm  = (const float*)d_in[15];
  const float* bm  = (const float*)d_in[16];
  const float* g_m = (const float*)d_in[17];
  const float* b_m = (const float*)d_in[18];

  // ---- workspace carve-up ----
  char* p = (char*)d_ws;
  auto take = [&](size_t sz) -> char* {
    char* r = p;
    p += (sz + 255) & ~(size_t)255;
    return r;
  };
  _Float16* wt_r  = (_Float16*)take((size_t)NPAD * 320 * 2);
  _Float16* wt_c  = (_Float16*)take((size_t)NPAD * 320 * 2);
  _Float16* wt_rs = (_Float16*)take((size_t)NPAD * 320 * 2);
  _Float16* wt_cs = (_Float16*)take((size_t)NPAD * 320 * 2);
  _Float16* wt_m  = (_Float16*)take((size_t)NPAD * 608 * 2);
  int*   deg   = (int*)take((size_t)4 * N * 4);       // [r_out, r_in, c_out, c_in]
  float* norms = (float*)take((size_t)4 * N * 4);     // same order
  _Float16* h_row = (_Float16*)take((size_t)N * 320 * 2);
  _Float16* h_col = (_Float16*)take((size_t)N * 320 * 2);
  float* agg_row = (float*)take((size_t)N * GN_D * 4);
  float* agg_col = (float*)take((size_t)N * GN_D * 4);
  // merged [N x 608] f16 aliases the (dead by then) agg buffers
  _Float16* merged = (_Float16*)agg_row;

  int* deg_r_out = deg;
  int* deg_r_in  = deg + N;
  int* deg_c_out = deg + 2 * N;
  int* deg_c_in  = deg + 3 * N;
  float* nr_out = norms;
  float* nr_in  = norms + N;
  float* nc_out = norms + 2 * N;
  float* nc_in  = norms + 3 * N;

  // ---- zero init (capture-safe async memsets) ----
  hipMemsetAsync(deg, 0, (size_t)4 * N * 4, stream);
  hipMemsetAsync(agg_row, 0, (size_t)N * GN_D * 4, stream);
  hipMemsetAsync(agg_col, 0, (size_t)N * GN_D * 4, stream);

  // ---- weight prep (transpose + pad to f16) ----
  {
    int tot320 = NPAD * 320, tot608 = NPAD * 608;
    prep_wt_kernel<<<(tot320 + 255) / 256, 256, 0, stream>>>(Wr,  wt_r,  GN_D, 320);
    prep_wt_kernel<<<(tot320 + 255) / 256, 256, 0, stream>>>(Wc,  wt_c,  GN_D, 320);
    prep_wt_kernel<<<(tot320 + 255) / 256, 256, 0, stream>>>(Wrs, wt_rs, GN_D, 320);
    prep_wt_kernel<<<(tot320 + 255) / 256, 256, 0, stream>>>(Wcs, wt_cs, GN_D, 320);
    prep_wt_kernel<<<(tot608 + 255) / 256, 256, 0, stream>>>(Wm,  wt_m,  2 * GN_D, 608);
  }

  // ---- degrees + norms ----
  degree_kernel<<<(E + 255) / 256, 256, 0, stream>>>(row_ed, E, deg_r_out, deg_r_in);
  degree_kernel<<<(E + 255) / 256, 256, 0, stream>>>(col_ed, E, deg_c_out, deg_c_in);
  norm_kernel<<<(4 * N + 255) / 256, 256, 0, stream>>>(deg, norms, 4 * N);

  // ---- edge scatter (wave per edge) ----
  {
    int blocks = (E * 32 + 255) / 256;
    scatter_kernel<<<blocks, 256, 0, stream>>>(x, row_ed, nr_out, agg_row, E);
    scatter_kernel<<<blocks, 256, 0, stream>>>(x, col_ed, nc_out, agg_col, E);
  }

  const int gblocks = N / BM;  // 3125

  // ---- GraphConv GEMMs: relu(norm_dst * agg @ W + b) -> f16 h[N x 320] ----
  gemm_fused_kernel<<<gblocks, 256, 0, stream>>>(
      agg_row, nr_in, nullptr, 0, GN_D, 320, wt_r, br,
      /*relu*/1, /*ln*/0, nullptr, nullptr,
      h_row, 320, 0, 0, nullptr);
  gemm_fused_kernel<<<gblocks, 256, 0, stream>>>(
      agg_col, nc_in, nullptr, 0, GN_D, 320, wt_c, bc,
      /*relu*/1, /*ln*/0, nullptr, nullptr,
      h_col, 320, 0, 0, nullptr);

  // ---- support GEMMs + LayerNorm -> merged f16 [N x 608] ----
  gemm_fused_kernel<<<gblocks, 256, 0, stream>>>(
      nullptr, nullptr, h_row, 320, 320, 320, wt_rs, brs,
      /*relu*/0, /*ln*/1, g_rs, b_rs,
      merged, 608, /*coff*/0, /*zero_hi*/0, nullptr);
  gemm_fused_kernel<<<gblocks, 256, 0, stream>>>(
      nullptr, nullptr, h_col, 320, 320, 320, wt_cs, bcs,
      /*relu*/0, /*ln*/1, g_cs, b_cs,
      merged, 608, /*coff*/300, /*zero_hi*/1, nullptr);

  // ---- merge GEMM + LayerNorm -> f32 d_out [N x 300] ----
  gemm_fused_kernel<<<gblocks, 256, 0, stream>>>(
      nullptr, nullptr, merged, 608, 608, 608, wt_m, bm,
      /*relu*/0, /*ln*/1, g_m, b_m,
      nullptr, 0, 0, 0, (float*)d_out);

  (void)in_sizes; (void)n_in; (void)out_size; (void)ws_size;
}
